// MCGConv_37022618092258
// MI455X (gfx1250) — matrix operand order repaired and checked by reading, not compiled, not run on hardware
//
#include <hip/hip_runtime.h>
#include <hip/hip_bf16.h>

#define DEVINL __device__ __forceinline__

typedef __attribute__((ext_vector_type(16))) __bf16 v16bf;
typedef __attribute__((ext_vector_type(8)))  float  v8f;

// ---- problem constants ----
constexpr int Bn = 8, Np = 4096, CIN = 64, COUT = 128;
constexpr int Sp = 1024, NSp = 32, Vp = 4;
constexpr float RAD2 = 0.2f * 0.2f;
constexpr int H1 = 64, H2 = 128;
constexpr float BNEPS = 1e-5f;
constexpr int GFC = 70;     // CIN + 6
constexpr int MIXK = 134;   // 2*CIN + 6
constexpr int MIXKP = 160;  // padded to 5*32

// ---- stats scratch layout (floats) ----
constexpr int ST_SUMB = 0, ST_SQB = 8, ST_INVB = 16,
              ST_BN1S = 24, ST_BN1Q = 88, ST_BN1M = 152, ST_BN1R = 216,
              ST_BN2S = 280, ST_BN2Q = 408, ST_BN2M = 536, ST_BN2R = 664,
              ST_TOT = 792;

DEVINL __bf16 to_bf16(float x) { return (__bf16)x; }
DEVINL float  gelu_f(float x)  { return 0.5f * x * (1.0f + erff(x * 0.7071067811865475f)); }

DEVINL float rng_uniform(unsigned x) {
  x ^= x >> 16; x *= 0x7feb352du; x ^= x >> 15; x *= 0x846ca68bu; x ^= x >> 16;
  float u = (float)(x >> 8) * (1.0f / 16777216.0f);
  return fmaxf(u, 1e-9f);
}

DEVINL v8f v8f_zero() { v8f z = {0.f,0.f,0.f,0.f,0.f,0.f,0.f,0.f}; return z; }

// A fragment: 16x32 bf16, rows striped per lane (ISA 7.12.2 16-bit A layout).
// lane m=lane&15 owns row m; hi=lane>>4 selects K offset +8 within each half.
DEVINL v16bf load_A_bf16(const __bf16* base, int ld, int rowBase, int kBase, int lane) {
  int m = lane & 15, hi = (lane >> 4) & 1;
  const __bf16* p = base + (size_t)(rowBase + m) * ld + kBase + hi * 8;
  v16bf a;
#pragma unroll
  for (int j = 0; j < 4; ++j) {
    a[2*j]     = p[2*j];
    a[2*j + 1] = p[2*j + 1];
    a[8 + 2*j] = p[16 + 2*j];
    a[9 + 2*j] = p[16 + 2*j + 1];
  }
  return a;
}

// B fragment: 32x16 bf16 built from weight W[o][k] row-major (B[k][n] = W[colBase+n][k]).
// lane n=lane&15 owns column n; hi=lane>>4 selects K offset +8 within each half (matches A).
DEVINL v16bf load_B_bf16(const __bf16* wgt, int ld, int colBase, int kBase, int lane) {
  int n = lane & 15, hi = (lane >> 4) & 1;
  const __bf16* p = wgt + (size_t)(colBase + n) * ld + kBase + hi * 8;
  v16bf b;
#pragma unroll
  for (int j = 0; j < 8; ++j) { b[j] = p[j]; b[8 + j] = p[16 + j]; }
  return b;
}

DEVINL v8f wmma_bf16(v16bf a, v16bf b, v8f c) {
  return __builtin_amdgcn_wmma_f32_16x16x32_bf16(false, a, false, b, (short)0, c, false, false);
}

// ================= K1: farthest point sampling =================
__global__ void fps_kernel(const float* __restrict__ xyz, int* __restrict__ idx,
                           float* __restrict__ nxyz) {
  int b = blockIdx.x, t = threadIdx.x;
  const float* X = xyz + (size_t)b * Np * 3;
  __shared__ float sv[256]; __shared__ int si[256];
  __shared__ float sp[3];   __shared__ int slast;
  float px[16], py[16], pz[16], mind[16];
#pragma unroll
  for (int i = 0; i < 16; ++i) {
    int n = t * 16 + i;
    px[i] = X[n*3]; py[i] = X[n*3+1]; pz[i] = X[n*3+2];
    mind[i] = 3.0e38f;
  }
  if (t == 0) slast = 0;
  __syncthreads();
  for (int it = 0; it < Sp; ++it) {
    int last = slast;
    if (t == 0) {
      idx[b * Sp + it] = last;
      float x = X[last*3], y = X[last*3+1], z = X[last*3+2];
      sp[0] = x; sp[1] = y; sp[2] = z;
      float* oz = nxyz + ((size_t)b * Sp + it) * 3;
      oz[0] = x; oz[1] = y; oz[2] = z;
    }
    __syncthreads();
    float cx = sp[0], cy = sp[1], cz = sp[2];
    float best = -1.f; int bi = t * 16;
#pragma unroll
    for (int i = 0; i < 16; ++i) {
      float dx = px[i]-cx, dy = py[i]-cy, dz = pz[i]-cz;
      float d = dx*dx + dy*dy + dz*dz;
      mind[i] = fminf(mind[i], d);
      if (mind[i] > best) { best = mind[i]; bi = t * 16 + i; }
    }
    sv[t] = best; si[t] = bi; __syncthreads();
    for (int o2 = 128; o2 > 0; o2 >>= 1) {
      if (t < o2) {
        float v2 = sv[t + o2]; int i2 = si[t + o2];
        if (v2 > sv[t] || (v2 == sv[t] && i2 < si[t])) { sv[t] = v2; si[t] = i2; }
      }
      __syncthreads();
    }
    if (t == 0) slast = si[0];
    __syncthreads();
  }
}

// ================= K2: ball query (first NS indices within radius) =================
__global__ void ball_kernel(const float* __restrict__ xyz, const float* __restrict__ nxyz,
                            int* __restrict__ gidx) {
  int gid = blockIdx.x * blockDim.x + threadIdx.x;
  if (gid >= Bn * Sp) return;
  int b = gid / Sp;
  const float* X = xyz + (size_t)b * Np * 3;
  float cx = nxyz[gid*3], cy = nxyz[gid*3+1], cz = nxyz[gid*3+2];
  int* g = gidx + (size_t)gid * NSp;
  int cnt = 0;
  for (int n = 0; n < Np && cnt < NSp; ++n) {
    float dx = X[n*3]-cx, dy = X[n*3+1]-cy, dz = X[n*3+2]-cz;
    if (dx*dx + dy*dy + dz*dz < RAD2) g[cnt++] = n;
  }
  for (; cnt < NSp; ++cnt) g[cnt] = -1;
}

// ================= K3: build gf (B,S,NS,70) + per-batch sum/sumsq =================
__global__ void gf_kernel(const float* __restrict__ xyz, const float* __restrict__ f,
                          const int* __restrict__ idxS, const int* __restrict__ gidx,
                          const float* __restrict__ nxyz, float* __restrict__ gf,
                          float* __restrict__ st) {
  int row = blockIdx.x * 256 + threadIdx.x;  // (b,s,n)
  int b  = row >> 15;                        // / (S*NS)
  int bs = row >> 5;                         // / NS
  int g = gidx[row];
  const float* nz = nxyz + (size_t)bs * 3;
  float nx = nz[0], ny = nz[1], nzv = nz[2];
  float gx = 0.f, gy = 0.f, gz = 0.f;
  if (g >= 0) { const float* p = xyz + ((size_t)b * Np + g) * 3; gx = p[0]; gy = p[1]; gz = p[2]; }
  float* o = gf + (size_t)row * GFC;
  float lsum = 0.f, lsq = 0.f;
  float v0[6] = { gx, gy, gz, gx - nx, gy - ny, gz - nzv };
#pragma unroll
  for (int k = 0; k < 6; ++k) { o[k] = v0[k]; lsum += v0[k]; lsq += v0[k] * v0[k]; }
  const float* Fs = f + ((size_t)b * Np + idxS[bs]) * CIN;
  const float* Fg = (g >= 0) ? (f + ((size_t)b * Np + g) * CIN) : nullptr;
  for (int c = 0; c < CIN; ++c) {
    float v = ((g >= 0) ? Fg[c] : 0.f) - Fs[c];
    o[6 + c] = v; lsum += v; lsq += v * v;
  }
  __shared__ float s1[256], s2[256];
  int t = threadIdx.x;
  s1[t] = lsum; s2[t] = lsq; __syncthreads();
  for (int o2 = 128; o2 > 0; o2 >>= 1) {
    if (t < o2) { s1[t] += s1[t + o2]; s2[t] += s2[t + o2]; }
    __syncthreads();
  }
  if (t == 0) { atomicAdd(&st[ST_SUMB + b], s1[0]); atomicAdd(&st[ST_SQB + b], s2[0]); }
}

__global__ void finalize_std_kernel(float* st) {
  int t = threadIdx.x;
  if (t < Bn) {
    float cnt = (float)(Sp * NSp * GFC);
    float sum = st[ST_SUMB + t], sq = st[ST_SQB + t];
    float var = (sq - sum * sum / cnt) / (cnt - 1.0f);
    st[ST_INVB + t] = 1.0f / (sqrtf(fmaxf(var, 0.f)) + 1e-5f);
  }
}

// ================= K5: mix GEMM (134->64) per (b,s), WMMA bf16 =================
__global__ void mix_kernel(const float* __restrict__ gf, const float* __restrict__ f,
                           const int* __restrict__ idxS, const float* __restrict__ aw,
                           const float* __restrict__ ab, const float* __restrict__ mw,
                           const float* __restrict__ mb, const float* __restrict__ st,
                           float* __restrict__ gfeat) {
  __shared__ __bf16 sA[32 * MIXKP];   // 10 KB
  __shared__ __bf16 sW[64 * MIXKP];   // 20 KB
  int bs = blockIdx.x, t = threadIdx.x;
  int b = bs / Sp;
  float inv = st[ST_INVB + b];
  const float* grow = gf + (size_t)bs * NSp * GFC;
  const float* Fs = f + ((size_t)b * Np + idxS[bs]) * CIN;
  for (int i = t; i < 64 * MIXKP; i += 256) {
    int o = i / MIXKP, k = i % MIXKP;
    sW[i] = (k < MIXK) ? to_bf16(mw[o * MIXK + k]) : to_bf16(0.f);
  }
  for (int i = t; i < 32 * MIXKP; i += 256) {
    int n = i / MIXKP, k = i % MIXKP;
    float v;
    if (k < GFC)       v = grow[n * GFC + k] * inv * aw[k] + ab[k];
    else if (k < MIXK) v = Fs[k - GFC];
    else               v = 0.f;
    sA[i] = to_bf16(v);
  }
  __syncthreads();
  int wv = t >> 5, lane = t & 31, hi = (lane >> 4) & 1;
  int rt = wv >> 2, ct = wv & 3;
  v8f acc = v8f_zero();
  for (int ks = 0; ks < 5; ++ks)
    acc = wmma_bf16(load_A_bf16(sA, MIXKP, rt * 16, ks * 32, lane),
                    load_B_bf16(sW, MIXKP, ct * 16, ks * 32, lane), acc);
  int col = ct * 16 + (lane & 15);
  float bb = mb[col];
  float* o = gfeat + (size_t)bs * NSp * 64;
#pragma unroll
  for (int r = 0; r < 8; ++r) { int row = rt * 16 + hi * 8 + r; o[row * 64 + col] = acc[r] + bb; }
}

// ================= K6: rel(13) -> h(64) GEMM + bn1 stats =================
__global__ void rel_kernel(const float* __restrict__ xyz, const int* __restrict__ cidx,
                           const float* __restrict__ gf, const float* __restrict__ w1,
                           const float* __restrict__ b1, __bf16* __restrict__ hbuf,
                           float* __restrict__ st) {
  __shared__ __bf16 sA[128 * 32];  // 8 KB
  __shared__ __bf16 sW[64 * 32];   // 4 KB
  __shared__ float smc[4][3];
  __shared__ float ssum[64], ssq[64];
  int bs = blockIdx.x, t = threadIdx.x;
  int b = bs / Sp;
  if (t < 12) { int v = t / 3, d = t % 3; smc[v][d] = xyz[((size_t)b * Np + cidx[v]) * 3 + d]; }
  if (t < 64) { ssum[t] = 0.f; ssq[t] = 0.f; }
  for (int i = t; i < 64 * 32; i += 256) {
    int o = i >> 5, k = i & 31;
    sW[i] = (k < 13) ? to_bf16(w1[o * 13 + k]) : to_bf16(0.f);
  }
  __syncthreads();
  for (int i = t; i < 128; i += 256) {
    int n = i >> 2, v = i & 3;
    const float* gr = gf + ((size_t)bs * NSp + n) * GFC;
    float gx = gr[0], gy = gr[1], gz = gr[2];
    float ax = gx - smc[v][0], ay = gy - smc[v][1], az = gz - smc[v][2];
    float eu = sqrtf(ax*ax + ay*ay + az*az);
    __bf16* r = sA + (size_t)i * 32;
    float vals[13] = { -ax,-ay,-az, ax,ay,az, eu, gx,gy,gz, smc[v][0],smc[v][1],smc[v][2] };
#pragma unroll
    for (int k = 0; k < 13; ++k) r[k] = to_bf16(vals[k]);
#pragma unroll
    for (int k = 13; k < 32; ++k) r[k] = to_bf16(0.f);
  }
  __syncthreads();
  int wv = t >> 5, lane = t & 31, hi = (lane >> 4) & 1;
  for (int ct = 0; ct < 4; ++ct) {
    v8f acc = v8f_zero();
    acc = wmma_bf16(load_A_bf16(sA, 32, wv * 16, 0, lane),
                    load_B_bf16(sW, 32, ct * 16, 0, lane), acc);
    int col = ct * 16 + (lane & 15);
    float bb = b1[col];
    float psum = 0.f, psq = 0.f;
#pragma unroll
    for (int r = 0; r < 8; ++r) {
      float val = acc[r] + bb;
      int row = wv * 16 + hi * 8 + r;
      hbuf[(size_t)(bs * 128 + row) * 64 + col] = to_bf16(val);
      psum += val; psq += val * val;
    }
    atomicAdd(&ssum[col], psum); atomicAdd(&ssq[col], psq);
  }
  __syncthreads();
  if (t < 64) { atomicAdd(&st[ST_BN1S + t], ssum[t]); atomicAdd(&st[ST_BN1Q + t], ssq[t]); }
}

__global__ void finalize_bn1_kernel(float* st) {
  int t = threadIdx.x;
  if (t < H1) {
    float cnt = (float)((long long)Bn * Sp * NSp * Vp);
    float mu = st[ST_BN1S + t] / cnt;
    float var = st[ST_BN1Q + t] / cnt - mu * mu;
    st[ST_BN1M + t] = mu;
    st[ST_BN1R + t] = rsqrtf(fmaxf(var, 0.f) + BNEPS);
  }
}

// ==== K8: bn+gelu -> w_jk GEMM -> gumbel softmax over V -> agg -> m2_w1 GEMM + bn2 stats ====
__global__ void gumbel_kernel(const __bf16* __restrict__ hbuf, const float* __restrict__ g1,
                              const float* __restrict__ be1, const float* __restrict__ w2,
                              const float* __restrict__ b2v, const float* __restrict__ gfeat,
                              const int* __restrict__ gidx, const float* __restrict__ w3,
                              const float* __restrict__ b3, __bf16* __restrict__ h2buf,
                              float* __restrict__ st) {
  __shared__ __bf16 sH[128 * 64];   // 16 KB
  __shared__ __bf16 sW2[64 * 64];   // 8 KB
  __shared__ __bf16 sAgg[32 * 64];  // 4 KB
  __shared__ __bf16 sW3[128 * 64];  // 16 KB
  __shared__ float ssum[128], ssq[128];
  int bs = blockIdx.x, t = threadIdx.x;
  const __bf16* hsrc = hbuf + (size_t)bs * 128 * 64;
  for (int i = t; i < 128 * 64; i += 256) {
    int c = i & 63;
    float x = (float)hsrc[i];
    x = (x - st[ST_BN1M + c]) * st[ST_BN1R + c] * g1[c] + be1[c];
    sH[i] = to_bf16(gelu_f(x));
  }
  for (int i = t; i < 64 * 64; i += 256)  sW2[i] = to_bf16(w2[i]);
  for (int i = t; i < 128 * 64; i += 256) sW3[i] = to_bf16(w3[i]);
  if (t < 128) { ssum[t] = 0.f; ssq[t] = 0.f; }
  __syncthreads();
  int wv = t >> 5, lane = t & 31, hi = (lane >> 4) & 1;
  const float* gfl = gfeat + (size_t)bs * NSp * 64;
  // w_jk rows 16*wv..16*wv+15 ; lane holds full V-groups for 2 n's -> lane-local softmax
  for (int ct = 0; ct < 4; ++ct) {
    v8f acc = v8f_zero();
    for (int ks = 0; ks < 2; ++ks)
      acc = wmma_bf16(load_A_bf16(sH, 64, wv * 16, ks * 32, lane),
                      load_B_bf16(sW2, 64, ct * 16, ks * 32, lane), acc);
    int col = ct * 16 + (lane & 15);
    float bb = b2v[col];
#pragma unroll
    for (int gp = 0; gp < 2; ++gp) {
      int n = 4 * wv + 2 * hi + gp;
      float gfv = gfl[n * 64 + col];
      float lg[4], z[4]; float mx = -3.0e38f;
#pragma unroll
      for (int v = 0; v < 4; ++v) {
        float wjk = acc[gp * 4 + v] + bb;
        lg[v] = gfv * wjk;
        unsigned id5 = (unsigned)((((bs * NSp + n) * Vp + v) << 6) + col);
        float u = rng_uniform(id5);
        z[v] = lg[v] - logf(-logf(u));   // logits + gumbel
        mx = fmaxf(mx, z[v]);
      }
      float den = 0.f, aggv = 0.f;
#pragma unroll
      for (int v = 0; v < 4; ++v) { float e = expf(z[v] - mx); den += e; z[v] = e; }
#pragma unroll
      for (int v = 0; v < 4; ++v) aggv += (z[v] / den) * lg[v];
      float agg = aggv + gfv;            // + group_skip
      if (gidx[bs * NSp + n] < 0) agg = 0.f;
      sAgg[n * 64 + col] = to_bf16(agg);
    }
  }
  __syncthreads();
  // h2 = agg(32x64) @ m2_w1^T(64x128) + b3, store bf16 + bn2 stats
  {
    int col = wv * 16 + (lane & 15);
    float bb = b3[col];
    float psum = 0.f, psq = 0.f;
    for (int rt = 0; rt < 2; ++rt) {
      v8f acc = v8f_zero();
      for (int ks = 0; ks < 2; ++ks)
        acc = wmma_bf16(load_A_bf16(sAgg, 64, rt * 16, ks * 32, lane),
                        load_B_bf16(sW3, 64, wv * 16, ks * 32, lane), acc);
#pragma unroll
      for (int r = 0; r < 8; ++r) {
        float v = acc[r] + bb;
        int row = rt * 16 + hi * 8 + r;
        h2buf[(size_t)(bs * 32 + row) * 128 + col] = to_bf16(v);
        psum += v; psq += v * v;
      }
    }
    atomicAdd(&ssum[col], psum); atomicAdd(&ssq[col], psq);
  }
  __syncthreads();
  if (t < 128) { atomicAdd(&st[ST_BN2S + t], ssum[t]); atomicAdd(&st[ST_BN2Q + t], ssq[t]); }
}

__global__ void finalize_bn2_kernel(float* st) {
  int t = threadIdx.x;
  if (t < H2) {
    float cnt = (float)((long long)Bn * Sp * NSp);
    float mu = st[ST_BN2S + t] / cnt;
    float var = st[ST_BN2Q + t] / cnt - mu * mu;
    st[ST_BN2M + t] = mu;
    st[ST_BN2R + t] = rsqrtf(fmaxf(var, 0.f) + BNEPS);
  }
}

// ==== K10: bn2 -> (m2_w2 GEMM) + (res GEMM) -> gelu -> max over NS -> new_f ====
__global__ void out_kernel(const __bf16* __restrict__ h2buf, const float* __restrict__ g2,
                           const float* __restrict__ be2, const float* __restrict__ w4,
                           const float* __restrict__ b4, const float* __restrict__ gfeat,
                           const float* __restrict__ w5, const float* __restrict__ b5,
                           const float* __restrict__ st, float* __restrict__ out_f) {
  __shared__ __bf16 sH2[32 * 128];   // 8 KB
  __shared__ __bf16 sGS[32 * 64];    // 4 KB
  __shared__ __bf16 sW4[128 * 128];  // 32 KB
  __shared__ __bf16 sW5[128 * 64];   // 16 KB
  int bs = blockIdx.x, t = threadIdx.x;
  const __bf16* hsrc = h2buf + (size_t)bs * 32 * 128;
  const float* gsrc = gfeat + (size_t)bs * 32 * 64;
  for (int i = t; i < 32 * 128; i += 256) {
    int c = i & 127;
    float x = (float)hsrc[i];
    x = (x - st[ST_BN2M + c]) * st[ST_BN2R + c] * g2[c] + be2[c];
    sH2[i] = to_bf16(x);
  }
  for (int i = t; i < 32 * 64; i += 256)   sGS[i] = to_bf16(gsrc[i]);
  for (int i = t; i < 128 * 128; i += 256) sW4[i] = to_bf16(w4[i]);
  for (int i = t; i < 128 * 64; i += 256)  sW5[i] = to_bf16(w5[i]);
  __syncthreads();
  int wv = t >> 5, lane = t & 31, hi = (lane >> 4) & 1;
  int col = wv * 16 + (lane & 15);
  float bias = b4[col] + b5[col];
  float pmax = -3.0e38f;
  for (int rt = 0; rt < 2; ++rt) {
    v8f ao = v8f_zero();
    for (int ks = 0; ks < 4; ++ks)
      ao = wmma_bf16(load_A_bf16(sH2, 128, rt * 16, ks * 32, lane),
                     load_B_bf16(sW4, 128, wv * 16, ks * 32, lane), ao);
    v8f ar = v8f_zero();
    for (int ks = 0; ks < 2; ++ks)
      ar = wmma_bf16(load_A_bf16(sGS, 64, rt * 16, ks * 32, lane),
                     load_B_bf16(sW5, 64, wv * 16, ks * 32, lane), ar);
#pragma unroll
    for (int r = 0; r < 8; ++r) pmax = fmaxf(pmax, gelu_f(ao[r] + ar[r] + bias));
  }
  pmax = fmaxf(pmax, __shfl_xor(pmax, 16, 32));  // lanes L and L+16 hold different rows
  if (hi == 0) out_f[(size_t)bs * COUT + col] = pmax;
}

// ================= launch =================
extern "C" void kernel_launch(void* const* d_in, const int* in_sizes, int n_in,
                              void* d_out, int out_size, void* d_ws, size_t ws_size,
                              hipStream_t stream) {
  (void)in_sizes; (void)n_in; (void)out_size; (void)ws_size;
  const float* xyz    = (const float*)d_in[0];
  const float* f      = (const float*)d_in[1];
  const int*   cidx   = (const int*)d_in[2];
  const float* aw     = (const float*)d_in[3];
  const float* ab     = (const float*)d_in[4];
  const float* mix_w  = (const float*)d_in[5];
  const float* mix_b  = (const float*)d_in[6];
  const float* m1_w1  = (const float*)d_in[7];
  const float* m1_b1  = (const float*)d_in[8];
  const float* m1_g1  = (const float*)d_in[9];
  const float* m1_be1 = (const float*)d_in[10];
  const float* m1_w2  = (const float*)d_in[11];
  const float* m1_b2  = (const float*)d_in[12];
  const float* m2_w1  = (const float*)d_in[13];
  const float* m2_b1  = (const float*)d_in[14];
  const float* m2_g1  = (const float*)d_in[15];
  const float* m2_be1 = (const float*)d_in[16];
  const float* m2_w2  = (const float*)d_in[17];
  const float* m2_b2  = (const float*)d_in[18];
  const float* res_w  = (const float*)d_in[19];
  const float* res_b  = (const float*)d_in[20];

  float* out_f   = (float*)d_out;
  float* out_xyz = out_f + (size_t)Bn * Sp * COUT;

  char* w = (char*)d_ws;
  size_t off = 0;
  auto take = [&](size_t bytes) -> char* {
    char* p = w + off;
    off = (off + bytes + 255) & ~(size_t)255;
    return p;
  };
  int*    ws_idx   = (int*)take((size_t)Bn * Sp * 4);
  int*    ws_gidx  = (int*)take((size_t)Bn * Sp * NSp * 4);
  float*  ws_gf    = (float*)take((size_t)Bn * Sp * NSp * GFC * 4);
  float*  ws_gfeat = (float*)take((size_t)Bn * Sp * NSp * CIN * 4);
  __bf16* ws_h     = (__bf16*)take((size_t)Bn * Sp * NSp * Vp * H1 * 2);
  __bf16* ws_h2    = (__bf16*)take((size_t)Bn * Sp * NSp * H2 * 2);
  float*  ws_st    = (float*)take((size_t)ST_TOT * 4);

  hipMemsetAsync(ws_st, 0, ST_TOT * 4, stream);

  fps_kernel<<<Bn, 256, 0, stream>>>(xyz, ws_idx, out_xyz);
  ball_kernel<<<(Bn * Sp + 255) / 256, 256, 0, stream>>>(xyz, out_xyz, ws_gidx);
  gf_kernel<<<(Bn * Sp * NSp) / 256, 256, 0, stream>>>(xyz, f, ws_idx, ws_gidx, out_xyz, ws_gf, ws_st);
  finalize_std_kernel<<<1, 32, 0, stream>>>(ws_st);
  mix_kernel<<<Bn * Sp, 256, 0, stream>>>(ws_gf, f, ws_idx, aw, ab, mix_w, mix_b, ws_st, ws_gfeat);
  rel_kernel<<<Bn * Sp, 256, 0, stream>>>(xyz, cidx, ws_gf, m1_w1, m1_b1, ws_h, ws_st);
  finalize_bn1_kernel<<<1, 64, 0, stream>>>(ws_st);
  gumbel_kernel<<<Bn * Sp, 256, 0, stream>>>(ws_h, m1_g1, m1_be1, m1_w2, m1_b2, ws_gfeat,
                                             ws_gidx, m2_w1, m2_b1, ws_h2, ws_st);
  finalize_bn2_kernel<<<1, 128, 0, stream>>>(ws_st);
  out_kernel<<<Bn * Sp, 256, 0, stream>>>(ws_h2, m2_g1, m2_be1, m2_w2, m2_b2, ws_gfeat,
                                          res_w, res_b, ws_st, out_f);
}
